// MyModel_86655260164219
// MI455X (gfx1250) — matrix operand order repaired and checked by reading, not compiled, not run on hardware
//
#include <hip/hip_runtime.h>
#include <hip/hip_bf16.h>

#define DEVI __device__ __forceinline__

typedef __attribute__((ext_vector_type(16))) __bf16 v16bf;
typedef __attribute__((ext_vector_type(8)))  float  v8f;

// ---- model dims (fixed by reference) ----
constexpr int   Nb = 2, S = 2560, C = 512, H = 8;
constexpr int   Dd = 5, T = 512, FF = 2048, HD = 64;
constexpr int   ND = Nb * Dd;           // 10 dilated groups
constexpr int   NS = Nb * S;            // 5120 tokens
constexpr float NEGV = -1e9f;

// ---- workspace layout (bytes) ----
constexpr size_t SZ_F32MAT = (size_t)NS * C * 4;        // 10 MB
constexpr size_t SZ_BFMAT  = (size_t)NS * C * 2;        //  5 MB
constexpr size_t OFF_X1  = 0;
constexpr size_t OFF_Y1  = OFF_X1 + SZ_F32MAT;
constexpr size_t OFF_Q   = OFF_Y1 + SZ_BFMAT;
constexpr size_t OFF_K   = OFF_Q  + SZ_BFMAT;
constexpr size_t OFF_VT  = OFF_K  + SZ_BFMAT;           // V transposed [bh][j][t]
constexpr size_t OFF_S   = OFF_VT + SZ_BFMAT;
constexpr size_t SZ_S    = (size_t)ND * H * T * T * 4;  // 80 MB f32 scores
constexpr size_t OFF_WB  = OFF_S  + SZ_S;               // bf16 softmax weights
constexpr size_t SZ_WB   = (size_t)ND * H * T * T * 2;  // 40 MB
constexpr size_t OFF_O   = OFF_WB + SZ_WB;
constexpr size_t OFF_X2  = OFF_O  + SZ_BFMAT;
constexpr size_t OFF_Y2  = OFF_X2 + SZ_F32MAT;
constexpr size_t OFF_HFF = OFF_Y2 + SZ_BFMAT;
constexpr size_t SZ_HFF  = (size_t)NS * FF * 2;         // 20 MB
// packed bf16 weights (B-fragment layout)
constexpr size_t OFF_PCONV = OFF_HFF + SZ_HFF;          // 3 taps x 512x512
constexpr size_t SZ_PW_CC  = (size_t)C * C * 2;         // 0.5 MB
constexpr size_t OFF_PQ  = OFF_PCONV + 3 * SZ_PW_CC;
constexpr size_t OFF_PK  = OFF_PQ + SZ_PW_CC;
constexpr size_t OFF_PV  = OFF_PK + SZ_PW_CC;
constexpr size_t OFF_PO  = OFF_PV + SZ_PW_CC;
constexpr size_t OFF_P1  = OFF_PO + SZ_PW_CC;           // 512x2048
constexpr size_t OFF_P2  = OFF_P1 + (size_t)C * FF * 2; // 2048x512

// ---- helpers ----
union Frag { v16bf bf; unsigned short u[16]; uint4 q[2]; };

DEVI unsigned short f2bf(float x) {            // round-to-nearest-even f32 -> bf16
  union { float f; unsigned int u; } c; c.f = x;
  unsigned int u = c.u;
  u += 0x7fffu + ((u >> 16) & 1u);
  return (unsigned short)(u >> 16);
}

// A fragment 16x32 bf16: row = lane&15, K-base 0/8 per lane half,
// elems[0..7]=kb+0..7, elems[8..15]=kb+16..23  -> two b128 loads
DEVI v16bf load_a_bf16v(const unsigned short* A, int lda, int m0, int k0, int lane) {
  Frag f;
  const unsigned short* p = A + (size_t)(m0 + (lane & 15)) * lda + k0 + ((lane & 16) ? 8 : 0);
  f.q[0] = *(const uint4*)p;
  f.q[1] = *(const uint4*)(p + 16);
  return f.bf;
}
// packed B: PB[((nt*KCS + kc32)*32 + lane)*16 + i] -> two b128 loads
DEVI v16bf load_b_packed(const unsigned short* PB, int kcs, int nt, int kc32, int lane) {
  Frag f;
  const unsigned short* p = PB + ((size_t)(nt * kcs + kc32) * 32 + lane) * 16;
  f.q[0] = *(const uint4*)p;
  f.q[1] = *(const uint4*)(p + 16);
  return f.bf;
}
DEVI v8f wmma_bf16(v16bf a, v16bf b, v8f acc) {
  return __builtin_amdgcn_wmma_f32_16x16x32_bf16(false, a, false, b, (short)0, acc, false, false);
}
DEVI float gelu_tanh(float x) {
  float x3 = x * x * x;
  return 0.5f * x * (1.f + tanhf(0.7978845608028654f * (x + 0.044715f * x3)));
}

// ---------- weight pre-pack: f32 (k stride ldk, n stride ldn) -> bf16 fragment layout ----------
__global__ void pack_b_kernel(const float* __restrict__ W, long ldk, long ldn,
                              unsigned short* __restrict__ Out) {
  int lane = threadIdx.x;                       // 32 threads
  int kc32 = blockIdx.x, nt = blockIdx.y;       // gridDim.x = Ktot/32
  int n  = nt * 16 + (lane & 15);
  int k0 = kc32 * 32 + ((lane & 16) ? 8 : 0);
  Frag f;
#pragma unroll
  for (int i = 0; i < 16; ++i)
    f.u[i] = f2bf(W[(size_t)(k0 + i + (i & 8)) * ldk + (size_t)n * ldn]);
  unsigned short* o = Out + ((size_t)(nt * gridDim.x + kc32) * 32 + lane) * 16;
  *(uint4*)o = f.q[0];
  *((uint4*)o + 1) = f.q[1];
}

// ---------- shared GEMM core: 1 A-load feeds 4 WMMAs ----------
template <int KTOT>
DEVI void gemm4(const unsigned short* A, int lda, int m0,
                const unsigned short* PB, int nt0, int lane, v8f acc[4]) {
  constexpr int KCS = KTOT / 32;
#pragma unroll 1
  for (int kc32 = 0; kc32 < KCS; ++kc32) {
    v16bf a = load_a_bf16v(A, lda, m0, kc32 * 32, lane);
#pragma unroll
    for (int j = 0; j < 4; ++j) {
      v16bf b = load_b_packed(PB, KCS, nt0 + j, kc32, lane);
      acc[j] = wmma_bf16(a, b, acc[j]);
    }
  }
}

// ---------- 1. causal dilated conv + gelu + residual ----------
__global__ void conv_kernel(const float* __restrict__ X, const float* __restrict__ Mask,
                            const unsigned short* __restrict__ PW,  // packed, per-tap
                            const float* __restrict__ Bc, float* __restrict__ X1) {
  int lane = threadIdx.x & 31, wid = threadIdx.x >> 5;
  int mt = blockIdx.y * 4 + wid;              // token tile 0..319
  int m0 = mt * 16, nt0 = blockIdx.x * 4;     // 4 out-channel tiles per wave
  int row = m0 + (lane & 15);
  int nb = row / S, s = row % S;
  int kb8 = (lane & 16) ? 8 : 0;
  v8f acc[4] = {};
#pragma unroll
  for (int tap = 0; tap < 3; ++tap) {
    int sp = s + (tap - 2) * 2;               // taps at s-4, s-2, s
    int spc = sp < 0 ? 0 : sp;
    const float* px = X + (size_t)(nb * S + spc) * C;
    float mk = sp < 0 ? 0.f : Mask[nb * S + sp];
    const unsigned short* PWt = PW + tap * ((size_t)C * C);
#pragma unroll 1
    for (int kc32 = 0; kc32 < C / 32; ++kc32) {
      Frag a;
      const float4* p0 = (const float4*)(px + kc32 * 32 + kb8);
      const float4* p1 = (const float4*)(px + kc32 * 32 + kb8 + 16);
      float4 u0 = p0[0], u1 = p0[1], v0 = p1[0], v1 = p1[1];
      a.u[0] = f2bf(u0.x * mk); a.u[1] = f2bf(u0.y * mk);
      a.u[2] = f2bf(u0.z * mk); a.u[3] = f2bf(u0.w * mk);
      a.u[4] = f2bf(u1.x * mk); a.u[5] = f2bf(u1.y * mk);
      a.u[6] = f2bf(u1.z * mk); a.u[7] = f2bf(u1.w * mk);
      a.u[8]  = f2bf(v0.x * mk); a.u[9]  = f2bf(v0.y * mk);
      a.u[10] = f2bf(v0.z * mk); a.u[11] = f2bf(v0.w * mk);
      a.u[12] = f2bf(v1.x * mk); a.u[13] = f2bf(v1.y * mk);
      a.u[14] = f2bf(v1.z * mk); a.u[15] = f2bf(v1.w * mk);
#pragma unroll
      for (int j = 0; j < 4; ++j) {
        v16bf b = load_b_packed(PWt, C / 32, nt0 + j, kc32, lane);
        acc[j] = wmma_bf16(a.bf, b, acc[j]);
      }
    }
  }
#pragma unroll
  for (int j = 0; j < 4; ++j)
#pragma unroll
    for (int r = 0; r < 8; ++r) {
      int n = lane & 15, m = r + ((lane & 16) ? 8 : 0);
      int g = m0 + m, oc = (nt0 + j) * 16 + n;
      float h = acc[j][r] + Bc[oc];
      X1[(size_t)g * C + oc] = X[(size_t)g * C + oc] + gelu_tanh(h) * Mask[g];
    }
}

// ---------- LayerNorm (optionally regrids n(t d)c -> (n d)t c) ----------
__global__ void ln_kernel(const float* __restrict__ Xin, const float* __restrict__ G,
                          const float* __restrict__ Bt, unsigned short* __restrict__ Yout,
                          int regrid) {
  __shared__ float rsum[256], rsq[256];
  int rowo = blockIdx.x;
  size_t rin;
  if (regrid) {
    int b = rowo / T, t = rowo % T;
    rin = (size_t)(b / Dd) * S + (size_t)t * Dd + (b % Dd);
  } else rin = rowo;
  const float* x = Xin + rin * C;
  int tid = threadIdx.x;
  float a0 = x[tid * 2], a1 = x[tid * 2 + 1];
  rsum[tid] = a0 + a1; rsq[tid] = a0 * a0 + a1 * a1;
  __syncthreads();
  for (int off = 128; off > 0; off >>= 1) {
    if (tid < off) { rsum[tid] += rsum[tid + off]; rsq[tid] += rsq[tid + off]; }
    __syncthreads();
  }
  float mean = rsum[0] * (1.f / C);
  float var  = rsq[0] * (1.f / C) - mean * mean;
  float rstd = rsqrtf(var + 1e-5f);
  unsigned short* y = Yout + (size_t)rowo * C;
  y[tid * 2]     = f2bf((a0 - mean) * rstd * G[tid * 2]     + Bt[tid * 2]);
  y[tid * 2 + 1] = f2bf((a1 - mean) * rstd * G[tid * 2 + 1] + Bt[tid * 2 + 1]);
}

// ---------- 3. fused Q/K/V projections (z selects); V stored transposed ----------
__global__ void gemm_qkv_kernel(const unsigned short* __restrict__ Y,
                                const unsigned short* __restrict__ PQ,
                                const unsigned short* __restrict__ PK,
                                const unsigned short* __restrict__ PV,
                                const float* __restrict__ Bq, const float* __restrict__ Bk,
                                const float* __restrict__ Bv,
                                unsigned short* __restrict__ Q, unsigned short* __restrict__ Kk,
                                unsigned short* __restrict__ Vt) {
  int lane = threadIdx.x & 31, wid = threadIdx.x >> 5;
  int mt = blockIdx.y * 4 + wid;
  int m0 = mt * 16, nt0 = blockIdx.x * 4;
  int z = blockIdx.z;
  const unsigned short* PB = (z == 0) ? PQ : (z == 1) ? PK : PV;
  const float* Bias = (z == 0) ? Bq : (z == 1) ? Bk : Bv;
  v8f acc[4] = {};
  gemm4<C>(Y, C, m0, PB, nt0, lane, acc);
#pragma unroll
  for (int j = 0; j < 4; ++j)
#pragma unroll
    for (int r = 0; r < 8; ++r) {
      int n = lane & 15, m = r + ((lane & 16) ? 8 : 0);
      int grow = m0 + m, ncol = (nt0 + j) * 16 + n;
      unsigned short val = f2bf(acc[j][r] + Bias[ncol]);
      if (z == 2) {                     // Vt[bh][j][t]
        int b = grow / T, t = grow % T;
        int h = ncol / HD, jj = ncol % HD;
        Vt[((size_t)(b * H + h) * HD + jj) * T + t] = val;
      } else {
        unsigned short* O = (z == 0) ? Q : Kk;
        O[(size_t)grow * C + ncol] = val;
      }
    }
}

// ---------- 4. scores = (q k^T)/8 with causal + key mask ----------
__global__ void scores_kernel(const unsigned short* __restrict__ Q,
                              const unsigned short* __restrict__ Km,
                              const float* __restrict__ Mask, float* __restrict__ Sws) {
  int lane = threadIdx.x & 31, wid = threadIdx.x >> 5;
  int kt = blockIdx.x * 4 + wid;      // key tile 0..31
  int qt = blockIdx.y;                // query tile 0..31
  int bh = blockIdx.z;                // 0..79
  int b = bh / H, h = bh % H;
  int q0 = qt * 16, t0 = kt * 16;
  float* Sbh = Sws + (size_t)bh * T * T;
  if (t0 > q0 + 15) {                 // fully above causal diagonal
#pragma unroll
    for (int r = 0; r < 8; ++r) {
      int n = lane & 15, m = r + ((lane & 16) ? 8 : 0);
      Sbh[(size_t)(q0 + m) * T + t0 + n] = NEGV;
    }
    return;
  }
  const unsigned short* Qb = Q + (size_t)b * T * C + h * HD;
  v8f acc = {};
#pragma unroll
  for (int kc = 0; kc < HD; kc += 32) {
    v16bf a = load_a_bf16v(Qb, C, q0, kc, lane);
    Frag fb;                           // B[j][t] = K[t][j] -> contiguous over j
    const unsigned short* pk = Km + (size_t)(b * T + t0 + (lane & 15)) * C
                                  + h * HD + kc + ((lane & 16) ? 8 : 0);
    fb.q[0] = *(const uint4*)pk;
    fb.q[1] = *(const uint4*)(pk + 16);
    acc = wmma_bf16(a, fb.bf, acc);
  }
  int nidx = b / Dd, d = b % Dd;
#pragma unroll
  for (int r = 0; r < 8; ++r) {
    int n = lane & 15, m = r + ((lane & 16) ? 8 : 0);
    int sq = q0 + m, tk = t0 + n;
    float mg = Mask[nidx * S + tk * Dd + d];
    float v = acc[r] * 0.125f;         // 1/sqrt(64)
    Sbh[(size_t)sq * T + tk] = (tk <= sq && mg > 0.f) ? v : NEGV;
  }
}

// ---------- 5. row softmax -> bf16 weights + scatter into R diagonal slots ----------
__global__ void softmax_kernel(const float* __restrict__ Sws, unsigned short* __restrict__ Wb,
                               float* __restrict__ R) {
  __shared__ float red[128];
  int rowi = blockIdx.x;              // 0..ND*H*T-1
  int s = rowi % T, bh = rowi / T;
  int h = bh % H, b = bh / H;
  int nidx = b / Dd, d = b % Dd;
  const float* row = Sws + (size_t)bh * T * T + (size_t)s * T;
  unsigned short* wrow = Wb + (size_t)bh * T * T + (size_t)s * T;
  int tid = threadIdx.x;
  float vals[4];
  float mx = NEGV;
#pragma unroll
  for (int i = 0; i < 4; ++i) { vals[i] = row[tid + 128 * i]; mx = fmaxf(mx, vals[i]); }
  red[tid] = mx; __syncthreads();
  for (int off = 64; off > 0; off >>= 1) {
    if (tid < off) red[tid] = fmaxf(red[tid], red[tid + off]);
    __syncthreads();
  }
  mx = red[0]; __syncthreads();
  float sum = 0.f;
#pragma unroll
  for (int i = 0; i < 4; ++i) { vals[i] = __expf(vals[i] - mx); sum += vals[i]; }
  red[tid] = sum; __syncthreads();
  for (int off = 64; off > 0; off >>= 1) {
    if (tid < off) red[tid] += red[tid + off];
    __syncthreads();
  }
  float inv = 1.f / red[0];
  float* Rb = R + (((size_t)nidx * H + h) * S + (size_t)s * Dd + d) * S + d;
#pragma unroll
  for (int i = 0; i < 4; ++i) {
    int t = tid + 128 * i;
    float w = vals[i] * inv;
    wrow[t] = f2bf(w);                // bf16 copy for o = w@v
    Rb[(size_t)t * Dd] = w;           // R[n,h, s*D+d, t*D+d]
  }
}

// ---------- 6. o = w @ v (causal-truncated K loop; 4 hd-tiles per wave) ----------
__global__ void av_kernel(const unsigned short* __restrict__ Wb,
                          const unsigned short* __restrict__ Vt,
                          unsigned short* __restrict__ O) {
  int lane = threadIdx.x & 31, wid = threadIdx.x >> 5;
  int mt = blockIdx.x * 4 + wid;       // q tile 0..31
  int bh = blockIdx.y;
  int b = bh / H, h = bh % H;
  int q0 = mt * 16;
  const unsigned short* Wrow = Wb + (size_t)bh * T * T;
  const unsigned short* Vb = Vt + (size_t)bh * HD * T;
  int kb16 = (lane & 16) ? 8 : 0;
  v8f acc[4] = {};
  int kend = (q0 + 47) & ~31; if (kend > T) kend = T;   // keys beyond q0+15 are zero
#pragma unroll 1
  for (int kc32 = 0; kc32 < kend / 32; ++kc32) {
    v16bf a = load_a_bf16v(Wrow, T, q0, kc32 * 32, lane);
#pragma unroll
    for (int j = 0; j < 4; ++j) {       // hd tiles
      Frag fb;                          // B[t][j] from Vt: contiguous over t
      const unsigned short* p = Vb + (size_t)(j * 16 + (lane & 15)) * T + kc32 * 32 + kb16;
      fb.q[0] = *(const uint4*)p;
      fb.q[1] = *(const uint4*)(p + 16);
      acc[j] = wmma_bf16(a, fb.bf, acc[j]);
    }
  }
#pragma unroll
  for (int j = 0; j < 4; ++j)
#pragma unroll
    for (int r = 0; r < 8; ++r) {
      int n = lane & 15, m = r + ((lane & 16) ? 8 : 0);
      O[(size_t)(b * T + q0 + m) * C + h * HD + j * 16 + n] = f2bf(acc[j][r]);
    }
}

// ---------- 7. out-proj + residual; un-grids (n d)t c -> n(t d)c ----------
__global__ void proj_kernel(const unsigned short* __restrict__ O,
                            const unsigned short* __restrict__ PWo,
                            const float* __restrict__ Bo, const float* __restrict__ X1,
                            float* __restrict__ X2) {
  int lane = threadIdx.x & 31, wid = threadIdx.x >> 5;
  int mt = blockIdx.y * 4 + wid;
  int m0 = mt * 16, nt0 = blockIdx.x * 4;
  v8f acc[4] = {};
  gemm4<C>(O, C, m0, PWo, nt0, lane, acc);
#pragma unroll
  for (int j = 0; j < 4; ++j)
#pragma unroll
    for (int r = 0; r < 8; ++r) {
      int n = lane & 15, m = r + ((lane & 16) ? 8 : 0);
      int grow = m0 + m, ncol = (nt0 + j) * 16 + n;
      int b = grow / T, t = grow % T;
      size_t urow = (size_t)(b / Dd) * S + (size_t)t * Dd + (b % Dd);
      X2[urow * C + ncol] = X1[urow * C + ncol] + acc[j][r] + Bo[ncol];
    }
}

// ---------- 9. ffn1: relu(y@w1 + b1) ----------
__global__ void ffn1_kernel(const unsigned short* __restrict__ Y2,
                            const unsigned short* __restrict__ PW1,
                            const float* __restrict__ B1, unsigned short* __restrict__ Hff) {
  int lane = threadIdx.x & 31, wid = threadIdx.x >> 5;
  int mt = blockIdx.y * 4 + wid;
  int m0 = mt * 16, nt0 = blockIdx.x * 4;
  v8f acc[4] = {};
  gemm4<C>(Y2, C, m0, PW1, nt0, lane, acc);
#pragma unroll
  for (int j = 0; j < 4; ++j)
#pragma unroll
    for (int r = 0; r < 8; ++r) {
      int n = lane & 15, m = r + ((lane & 16) ? 8 : 0);
      int ncol = (nt0 + j) * 16 + n;
      Hff[(size_t)(m0 + m) * FF + ncol] = f2bf(fmaxf(acc[j][r] + B1[ncol], 0.f));
    }
}

// ---------- 10. ffn2 + residual -> final x ----------
__global__ void ffn2_kernel(const unsigned short* __restrict__ Hff,
                            const unsigned short* __restrict__ PW2,
                            const float* __restrict__ B2, const float* __restrict__ X2,
                            float* __restrict__ Out) {
  int lane = threadIdx.x & 31, wid = threadIdx.x >> 5;
  int mt = blockIdx.y * 4 + wid;
  int m0 = mt * 16, nt0 = blockIdx.x * 4;
  v8f acc[4] = {};
  gemm4<FF>(Hff, FF, m0, PW2, nt0, lane, acc);
#pragma unroll
  for (int j = 0; j < 4; ++j)
#pragma unroll
    for (int r = 0; r < 8; ++r) {
      int n = lane & 15, m = r + ((lane & 16) ? 8 : 0);
      size_t idx = (size_t)(m0 + m) * C + (nt0 + j) * 16 + n;
      Out[idx] = X2[idx] + acc[j][r] + B2[(nt0 + j) * 16 + n];
    }
}

// ---------- zero-fill R (dominant BW cost: 420 MB) ----------
__global__ void zero_kernel(float4* __restrict__ p, long n4) {
  long i = (long)blockIdx.x * blockDim.x + threadIdx.x;
  long stride = (long)gridDim.x * blockDim.x;
  float4 z = make_float4(0.f, 0.f, 0.f, 0.f);
  for (; i < n4; i += stride) p[i] = z;
}

extern "C" void kernel_launch(void* const* d_in, const int* in_sizes, int n_in,
                              void* d_out, int out_size, void* d_ws, size_t ws_size,
                              hipStream_t stream) {
  const float* x      = (const float*)d_in[0];
  const float* mask   = (const float*)d_in[1];
  const float* conv_w = (const float*)d_in[2];
  const float* conv_b = (const float*)d_in[3];
  const float* ln1_g  = (const float*)d_in[4];
  const float* ln1_b  = (const float*)d_in[5];
  const float* wq = (const float*)d_in[6];  const float* bq = (const float*)d_in[7];
  const float* wk = (const float*)d_in[8];  const float* bk = (const float*)d_in[9];
  const float* wv = (const float*)d_in[10]; const float* bv = (const float*)d_in[11];
  const float* wo = (const float*)d_in[12]; const float* bo = (const float*)d_in[13];
  const float* ln2_g = (const float*)d_in[14]; const float* ln2_b = (const float*)d_in[15];
  const float* w1 = (const float*)d_in[16]; const float* b1 = (const float*)d_in[17];
  const float* w2 = (const float*)d_in[18]; const float* b2 = (const float*)d_in[19];

  char* ws = (char*)d_ws;
  float*          X1  = (float*)(ws + OFF_X1);
  unsigned short* Y1  = (unsigned short*)(ws + OFF_Y1);
  unsigned short* Q   = (unsigned short*)(ws + OFF_Q);
  unsigned short* Kk  = (unsigned short*)(ws + OFF_K);
  unsigned short* Vt  = (unsigned short*)(ws + OFF_VT);
  float*          Sws = (float*)(ws + OFF_S);
  unsigned short* Wb  = (unsigned short*)(ws + OFF_WB);
  unsigned short* O   = (unsigned short*)(ws + OFF_O);
  float*          X2  = (float*)(ws + OFF_X2);
  unsigned short* Y2  = (unsigned short*)(ws + OFF_Y2);
  unsigned short* Hff = (unsigned short*)(ws + OFF_HFF);
  unsigned short* Pconv = (unsigned short*)(ws + OFF_PCONV);
  unsigned short* PQ  = (unsigned short*)(ws + OFF_PQ);
  unsigned short* PK  = (unsigned short*)(ws + OFF_PK);
  unsigned short* PV  = (unsigned short*)(ws + OFF_PV);
  unsigned short* PO  = (unsigned short*)(ws + OFF_PO);
  unsigned short* P1  = (unsigned short*)(ws + OFF_P1);
  unsigned short* P2  = (unsigned short*)(ws + OFF_P2);

  float* OutX = (float*)d_out;
  float* OutR = OutX + (size_t)Nb * S * C;

  // zero R (420 MB) — dominant cost; runs while packs execute on other WGPs? (stream-ordered,
  // but launched first so it overlaps nothing it depends on)
  zero_kernel<<<4096, 256, 0, stream>>>((float4*)OutR, (long)Nb * H * S * S / 4);

  // weight pre-pack into WMMA B-fragment bf16 layout
  for (int tap = 0; tap < 3; ++tap)   // B[ic][oc] = conv_w[(oc*C+ic)*3+tap]
    pack_b_kernel<<<dim3(C / 32, C / 16), 32, 0, stream>>>(
        conv_w + tap, 3L, (long)C * 3, Pconv + (size_t)tap * C * C);
  pack_b_kernel<<<dim3(C / 32, C / 16), 32, 0, stream>>>(wq, (long)C, 1L, PQ);
  pack_b_kernel<<<dim3(C / 32, C / 16), 32, 0, stream>>>(wk, (long)C, 1L, PK);
  pack_b_kernel<<<dim3(C / 32, C / 16), 32, 0, stream>>>(wv, (long)C, 1L, PV);
  pack_b_kernel<<<dim3(C / 32, C / 16), 32, 0, stream>>>(wo, (long)C, 1L, PO);
  pack_b_kernel<<<dim3(C / 32, FF / 16), 32, 0, stream>>>(w1, (long)FF, 1L, P1);
  pack_b_kernel<<<dim3(FF / 32, C / 16), 32, 0, stream>>>(w2, (long)C, 1L, P2);

  conv_kernel<<<dim3(C / 64, NS / 64), 128, 0, stream>>>(x, mask, Pconv, conv_b, X1);
  ln_kernel<<<ND * T, 256, 0, stream>>>(X1, ln1_g, ln1_b, Y1, 1);
  gemm_qkv_kernel<<<dim3(C / 64, (ND * T) / 64, 3), 128, 0, stream>>>(
      Y1, PQ, PK, PV, bq, bk, bv, Q, Kk, Vt);
  scores_kernel<<<dim3(T / 64, T / 16, ND * H), 128, 0, stream>>>(Q, Kk, mask, Sws);
  softmax_kernel<<<ND * H * T, 128, 0, stream>>>(Sws, Wb, OutR);
  av_kernel<<<dim3(T / 64, ND * H), 128, 0, stream>>>(Wb, Vt, O);
  proj_kernel<<<dim3(C / 64, (ND * T) / 64), 128, 0, stream>>>(O, PO, bo, X1, X2);
  ln_kernel<<<NS, 256, 0, stream>>>(X2, ln2_g, ln2_b, Y2, 0);
  ffn1_kernel<<<dim3(FF / 64, NS / 64), 128, 0, stream>>>(Y2, P1, b1, Hff);
  ffn2_kernel<<<dim3(C / 64, NS / 64), 128, 0, stream>>>(Hff, P2, b2, X2, OutX);
}
